// GraphSAGENet_20804821581834
// MI455X (gfx1250) — matrix operand order repaired and checked by reading, not compiled, not run on hardware
//
#include <hip/hip_runtime.h>
#include <hip/hip_bf16.h>

// ---------------------------------------------------------------------------
// GraphSAGE 2-layer inference for MI455X (gfx1250, wave32).
//  - aggregation: f32 global atomics (accumulator lives in 192MB L2)
//  - tile staging: GLOBAL_LOAD_ASYNC_TO_LDS_B128 (ASYNCcnt) straight to LDS
//  - GEMMs: v_wmma_f32_16x16x32_bf16, f32 accumulate
// ---------------------------------------------------------------------------

typedef __attribute__((ext_vector_type(16))) __bf16 v16bf;
typedef __attribute__((ext_vector_type(8)))  float  v8f;
typedef __bf16 bf16_t;

#define N_NODES 50000
#define N_EDGES 800000
#define NF      128
#define HC      128
#define NC      40
#define NCP     48      // NC padded to 3 x 16 WMMA tiles
#define LPAD    8       // LDS row pad: keeps 16B alignment for B128 async writes

// workspace layout (bytes); total ~25.95 MB
#define OFF_CNT 0u
#define OFF_AGG (256u * 1024u)                          // 50000 f32 counts, padded
#define OFF_WL1 (OFF_AGG + (unsigned)N_NODES * HC * 4u) // 25.6MB f32 accumulator
#define OFF_WR1 (OFF_WL1 + (unsigned)HC * NF * 2u)
#define OFF_WL2 (OFF_WR1 + (unsigned)HC * NF * 2u)
#define OFF_WR2 (OFF_WL2 + (unsigned)NCP * HC * 2u)

// ---------------------------------------------------------------- utilities
__global__ void zero_f128(float4* __restrict__ p, int n4) {
  int i = blockIdx.x * blockDim.x + threadIdx.x;
  if (i < n4) p[i] = make_float4(0.f, 0.f, 0.f, 0.f);
}

// convert f32 weight [rows x cols] to bf16 [prows x cols], zero-padding rows
__global__ void cvt_w(const float* __restrict__ w, bf16_t* __restrict__ o,
                      int rows, int cols, int prows) {
  int idx = blockIdx.x * blockDim.x + threadIdx.x;
  if (idx >= prows * cols) return;
  int r = idx / cols;
  o[idx] = (r < rows) ? (bf16_t)w[idx] : (bf16_t)0.0f;
}

// one thread = (edge, 4 consecutive features); 32 threads cover one edge row.
// Lane 0 of each edge also bumps the degree count when count_flag is set.
__global__ __launch_bounds__(256)
void scatter_feat(const float* __restrict__ feat, const int* __restrict__ src,
                  const int* __restrict__ dst, float* __restrict__ agg,
                  float* __restrict__ cnt, int relu_flag, int count_flag) {
  int tid = blockIdx.x * 256 + threadIdx.x;         // < 25.6M
  int e = tid >> 5;
  if (e >= N_EDGES) return;
  int f = (tid & 31) << 2;
  int s = src[e], d = dst[e];
  if ((tid & 31) == 0) {
    if (count_flag) atomicAdd(&cnt[d], 1.0f);
    if (e + 8 < N_EDGES)                            // pull next rows toward L2
      __builtin_prefetch(feat + (size_t)src[e + 8] * HC, 0, 0);
  }
  float4 v = *(const float4*)(feat + (size_t)s * HC + f);
  if (relu_flag) {
    v.x = fmaxf(v.x, 0.f); v.y = fmaxf(v.y, 0.f);
    v.z = fmaxf(v.z, 0.f); v.w = fmaxf(v.w, 0.f);
  }
  float* o = agg + (size_t)d * HC + f;
  atomicAdd(o + 0, v.x); atomicAdd(o + 1, v.y);
  atomicAdd(o + 2, v.z); atomicAdd(o + 3, v.w);
}

// ------------------------------------------------------------ WMMA helpers
// A-matrix (16-bit, 16x32 MxK) per-lane pack: m = lane&15;
//   lanes 0-15 hold K 0-15, lanes 16-31 hold K 16-31 (contiguous, same order as B).
// B-matrix (32x16 KxN): lane holds column n = lane&15; elements i = 16
//   consecutive K starting at (lane<16 ? 0 : 16)  -> contiguous weight row.
// C/D (16x16 f32): vgpr r -> row r + 8*(lane>=16), col lane&15.

// async global -> LDS, 16 bytes per lane (VDST = LDS byte address VGPR)
__device__ __forceinline__ void async_copy16(unsigned lds_addr, const void* gptr) {
  unsigned long long ga = (unsigned long long)gptr;
  asm volatile("global_load_async_to_lds_b128 %0, %1, off"
               :: "v"(lds_addr), "v"(ga) : "memory");
}
__device__ __forceinline__ void async_wait0() {
  asm volatile("s_wait_asynccnt 0x0" ::: "memory");
}

template <bool RELU, bool SCALE>
__device__ __forceinline__ v16bf pack_a_row(const float* row, int kA, float s) {
  v16bf a;
#pragma unroll
  for (int j = 0; j < 8; ++j) {
    int k = kA + 2 * j;
    float2 v = *(const float2*)(row + k);
    float e0 = v.x, e1 = v.y;
    if (SCALE) { e0 *= s; e1 *= s; }
    if (RELU)  { e0 = fmaxf(e0, 0.f); e1 = fmaxf(e1, 0.f); }
    a[2 * j]     = (bf16_t)e0;
    a[2 * j + 1] = (bf16_t)e1;
  }
  return a;
}

__device__ __forceinline__ v16bf load_b_row(const bf16_t* p) {
  v16bf b;
#pragma unroll
  for (int i = 0; i < 16; ++i) b[i] = p[i];
  return b;
}

// --------------------------------------------------- layer 1: emb [N x 128]
__global__ __launch_bounds__(256)
void sage1_wmma(const float* __restrict__ x, const float* __restrict__ agg,
                const float* __restrict__ cnt, const bf16_t* __restrict__ wl,
                const bf16_t* __restrict__ wr, const float* __restrict__ bias,
                float* __restrict__ emb) {
  __shared__ float sM[16][NF + LPAD];   // raw agg rows (scaled at pack time)
  __shared__ float sX[16][NF + LPAD];   // x rows
  const int node0 = blockIdx.x * 16;
  for (int i = threadIdx.x; i < (16 * NF) / 4; i += 256) {
    int r = i >> 5, c = (i & 31) << 2;
    async_copy16((unsigned)(size_t)&sM[r][c], agg + (size_t)(node0 + r) * NF + c);
    async_copy16((unsigned)(size_t)&sX[r][c], x   + (size_t)(node0 + r) * NF + c);
  }
  async_wait0();
  __syncthreads();

  const int lane = threadIdx.x & 31;
  const int wave = threadIdx.x >> 5;          // 8 waves = 8 n-tiles of 16
  const int m    = lane & 15;
  const int hi   = lane >> 4;
  const int n    = wave * 16 + m;             // output column (0..127)

  float cv = cnt[node0 + m];                  // this lane's A row only
  cv = cv < 1.f ? 1.f : cv;
  const float rs = 1.0f / cv;                 // fold mean divide into cvt

  v8f acc;
#pragma unroll
  for (int r = 0; r < 8; ++r) acc[r] = 0.0f;

#pragma unroll
  for (int kk = 0; kk < NF / 32; ++kk) {
    const int kA = kk * 32 + hi * 16;
    const int kB = kk * 32 + hi * 16;
    v16bf aM = pack_a_row<false, true >(&sM[m][0], kA, rs);
    v16bf aX = pack_a_row<false, false>(&sX[m][0], kA, 1.f);
    v16bf bL = load_b_row(wl + (size_t)n * NF + kB);
    v16bf bR = load_b_row(wr + (size_t)n * NF + kB);
    acc = __builtin_amdgcn_wmma_f32_16x16x32_bf16(false, aM, false, bL,
                                                  (short)0, acc, false, false);
    acc = __builtin_amdgcn_wmma_f32_16x16x32_bf16(false, aX, false, bR,
                                                  (short)0, acc, false, false);
  }
  const float bv = bias[n];
#pragma unroll
  for (int r = 0; r < 8; ++r) {
    int mo = r + hi * 8;
    emb[(size_t)(node0 + mo) * HC + n] = acc[r] + bv;
  }
}

// ------------------------------------------------- layer 2: logits [N x 40]
__global__ __launch_bounds__(96)
void sage2_wmma(const float* __restrict__ emb, const float* __restrict__ agg,
                const float* __restrict__ cnt, const bf16_t* __restrict__ wl,
                const bf16_t* __restrict__ wr, const float* __restrict__ bias,
                float* __restrict__ logits) {
  __shared__ float sM[16][HC + LPAD];   // raw agg rows
  __shared__ float sH[16][HC + LPAD];   // raw emb rows (ReLU at pack time)
  const int node0 = blockIdx.x * 16;
  for (int i = threadIdx.x; i < (16 * HC) / 4; i += 96) {
    int r = i >> 5, c = (i & 31) << 2;
    async_copy16((unsigned)(size_t)&sM[r][c], agg + (size_t)(node0 + r) * HC + c);
    async_copy16((unsigned)(size_t)&sH[r][c], emb + (size_t)(node0 + r) * HC + c);
  }
  async_wait0();
  __syncthreads();

  const int lane = threadIdx.x & 31;
  const int wave = threadIdx.x >> 5;              // 3 waves = 3 n-tiles (48)
  const int m    = lane & 15;
  const int hi   = lane >> 4;
  const int n    = wave * 16 + m;                 // 0..47 (40..47 padded)

  float cv = cnt[node0 + m];
  cv = cv < 1.f ? 1.f : cv;
  const float rs = 1.0f / cv;

  v8f acc;
#pragma unroll
  for (int r = 0; r < 8; ++r) acc[r] = 0.0f;

#pragma unroll
  for (int kk = 0; kk < HC / 32; ++kk) {
    const int kA = kk * 32 + hi * 16;
    const int kB = kk * 32 + hi * 16;
    v16bf aM = pack_a_row<true,  true >(&sM[m][0], kA, rs);  // relu(mean of h): h>=0 so relu(sum)/c == sum(relu)/c is NOT needed -- relu applied pre-aggregation in scatter; here scale only is fine, relu is a no-op on already-relu'd sums
    v16bf aH = pack_a_row<true,  false>(&sH[m][0], kA, 1.f); // relu(emb) = h
    v16bf bL = load_b_row(wl + (size_t)n * HC + kB);
    v16bf bR = load_b_row(wr + (size_t)n * HC + kB);
    acc = __builtin_amdgcn_wmma_f32_16x16x32_bf16(false, aM, false, bL,
                                                  (short)0, acc, false, false);
    acc = __builtin_amdgcn_wmma_f32_16x16x32_bf16(false, aH, false, bR,
                                                  (short)0, acc, false, false);
  }
  if (n < NC) {
    const float bv = bias[n];
#pragma unroll
    for (int r = 0; r < 8; ++r) {
      int mo = r + hi * 8;
      logits[(size_t)(node0 + mo) * NC + n] = acc[r] + bv;
    }
  }
}

// ---------------------------------------------------------------- launcher
extern "C" void kernel_launch(void* const* d_in, const int* in_sizes, int n_in,
                              void* d_out, int out_size, void* d_ws, size_t ws_size,
                              hipStream_t stream) {
  const float* x   = (const float*)d_in[0];
  const int*   ei  = (const int*)d_in[1];
  const float* Wl1 = (const float*)d_in[2];
  const float* Wr1 = (const float*)d_in[3];
  const float* b1  = (const float*)d_in[4];
  const float* Wl2 = (const float*)d_in[5];
  const float* Wr2 = (const float*)d_in[6];
  const float* b2  = (const float*)d_in[7];
  const int* src = ei;
  const int* dst = ei + N_EDGES;

  float* logits = (float*)d_out;                            // [N, 40]
  float* emb    = (float*)d_out + (size_t)N_NODES * NC;     // [N, 128]

  char* ws = (char*)d_ws;
  float*  cnt  = (float*)(ws + OFF_CNT);
  float*  agg  = (float*)(ws + OFF_AGG);
  bf16_t* wl1b = (bf16_t*)(ws + OFF_WL1);
  bf16_t* wr1b = (bf16_t*)(ws + OFF_WR1);
  bf16_t* wl2b = (bf16_t*)(ws + OFF_WL2);
  bf16_t* wr2b = (bf16_t*)(ws + OFF_WR2);

  const int NB16 = N_NODES / 16;                        // 3125, exact
  const int ZB4  = (N_NODES * HC / 4 + 255) / 256;      // 6250
  const int SB   = (N_EDGES * 32) / 256;                // 100000

  // init workspace (d_ws is poisoned; zero what atomics accumulate into)
  zero_f128<<<(N_NODES / 4 + 255) / 256, 256, 0, stream>>>((float4*)cnt, N_NODES / 4);
  zero_f128<<<ZB4, 256, 0, stream>>>((float4*)agg, N_NODES * HC / 4);
  cvt_w<<<(HC * NF + 255) / 256, 256, 0, stream>>>(Wl1, wl1b, HC, NF, HC);
  cvt_w<<<(HC * NF + 255) / 256, 256, 0, stream>>>(Wr1, wr1b, HC, NF, HC);
  cvt_w<<<(NCP * HC + 255) / 256, 256, 0, stream>>>(Wl2, wl2b, NC, HC, NCP);
  cvt_w<<<(NCP * HC + 255) / 256, 256, 0, stream>>>(Wr2, wr2b, NC, HC, NCP);

  // layer 1 (degree counting fused into the scatter)
  scatter_feat<<<SB, 256, 0, stream>>>(x, src, dst, agg, cnt, 0, 1);
  sage1_wmma<<<NB16, 256, 0, stream>>>(x, agg, cnt, wl1b, wr1b, b1, emb);

  // layer 2 (reuse agg)
  zero_f128<<<ZB4, 256, 0, stream>>>((float4*)agg, N_NODES * HC / 4);
  scatter_feat<<<SB, 256, 0, stream>>>(emb, src, dst, agg, cnt, 1, 0);
  sage2_wmma<<<NB16, 96, 0, stream>>>(emb, agg, cnt, wl2b, wr2b, b2, logits);
}
